// RestormerCN_36764920054563
// MI455X (gfx1250) — matrix operand order repaired and checked by reading, not compiled, not run on hardware
//
#include <hip/hip_runtime.h>

// ---------------------------------------------------------------------------
// RestormerCN channel-attention block for gfx1250 (MI455X, wave32, WMMA)
// All GEMM-shaped work uses v_wmma_f32_16x16x32_bf16.
// Round 2: 4x M-tile register blocking on the big GEMMs (B-fragment reuse),
//          8-way K-split on the attention-logits GEMM for occupancy.
// ---------------------------------------------------------------------------

typedef __attribute__((ext_vector_type(8)))  __bf16 v8bf;
typedef __attribute__((ext_vector_type(16))) __bf16 v16bf;
typedef __attribute__((ext_vector_type(8)))  float  v8f;

#define BN   8
#define CN   192
#define HN   128
#define WN   128
#define SN   16384      // HN*WN
#define C3N  576        // 3*CN
#define NHD  4
#define CHN  48         // CN/NHD
#define EPSN 1e-12f
#define KSPL 8          // K-split factor for attention logits

static __device__ __forceinline__ v16bf ld16(const __bf16* p0, const __bf16* p1) {
  v8bf lo = *(const v8bf*)p0;
  v8bf hi = *(const v8bf*)p1;
  v16bf r;
#pragma unroll
  for (int i = 0; i < 8; ++i) { r[i] = lo[i]; r[i + 8] = hi[i]; }
  return r;
}

static __device__ __forceinline__ v16bf zero16() {
  v16bf r;
#pragma unroll
  for (int i = 0; i < 16; ++i) r[i] = (__bf16)0.0f;
  return r;
}

static __device__ __forceinline__ v8f zero8() {
  v8f r;
#pragma unroll
  for (int i = 0; i < 8; ++i) r[i] = 0.0f;
  return r;
}

#define WMMA_BF16(A, Bm, Cacc) \
  __builtin_amdgcn_wmma_f32_16x16x32_bf16(false, (A), false, (Bm), (short)0, (Cacc), false, false)

// ---------------------------------------------------------------------------
// 1) f32 [b][c][s]  ->  bf16 [b][s][c]   (LDS-tiled transpose + convert)
// ---------------------------------------------------------------------------
__global__ void k_transpose_bf16(const float* __restrict__ in, __bf16* __restrict__ out) {
  __shared__ float tile[32][33];
  const int b  = blockIdx.z;
  const int s0 = blockIdx.x * 32;
  const int c0 = blockIdx.y * 32;
  const int tx = threadIdx.x, ty = threadIdx.y;
#pragma unroll
  for (int j = 0; j < 32; j += 8)
    tile[ty + j][tx] = in[((size_t)b * CN + (c0 + ty + j)) * SN + (s0 + tx)];
  __syncthreads();
#pragma unroll
  for (int j = 0; j < 32; j += 8)
    out[((size_t)b * SN + (s0 + ty + j)) * CN + (c0 + tx)] = (__bf16)tile[tx][ty + j];
}

// ---------------------------------------------------------------------------
// 2) flat f32 -> bf16 weight convert, and 3x3 weight permute (co,ci,t)->(t,co,ci)
// ---------------------------------------------------------------------------
__global__ void k_cvt(const float* __restrict__ src, __bf16* __restrict__ dst, int n) {
  int i = blockIdx.x * blockDim.x + threadIdx.x;
  if (i < n) dst[i] = (__bf16)src[i];
}

__global__ void k_cvt_w3t(const float* __restrict__ src, __bf16* __restrict__ dst) {
  int o = blockIdx.x * blockDim.x + threadIdx.x;
  if (o >= 9 * CN * CN) return;
  int t   = o / (CN * CN);
  int rem = o % (CN * CN);
  int co  = rem / CN;
  int ci  = rem % CN;
  dst[o] = (__bf16)src[((size_t)co * CN + ci) * 9 + t];
}

// ---------------------------------------------------------------------------
// 3) 1x1 conv as GEMM:  OutT[b][s][m] = sum_k W[m][k] * InT[b][s][k],  K = CN
//    one wave = 64x16 output (4 M-tiles share each B fragment), 24 WMMA.
// ---------------------------------------------------------------------------
__global__ void k_gemm_nt(const __bf16* __restrict__ Wt, const __bf16* __restrict__ InT,
                          __bf16* __restrict__ OutT, int M, int Mblocks) {
  const int wid  = blockIdx.x * (blockDim.x >> 5) + (threadIdx.x >> 5);
  const int lane = threadIdx.x & 31;
  const int Ntiles = SN / 16;
  if (wid >= BN * Mblocks * Ntiles) return;
  const int b  = wid / (Mblocks * Ntiles);
  const int r  = wid % (Mblocks * Ntiles);
  const int mb = r / Ntiles, nt = r % Ntiles;
  const int m0 = mb * 64, s0 = nt * 16;
  const int l15  = lane & 15;
  const int hi   = lane >> 4;
  const int aoff = hi * 8, boff = hi * 8, base = hi * 8;

  const __bf16* brow = InT + ((size_t)b * SN + (s0 + l15)) * CN;
  const __bf16* arow = Wt + (size_t)(m0 + l15) * CN;

  v8f acc[4];
#pragma unroll
  for (int mi = 0; mi < 4; ++mi) acc[mi] = zero8();

#pragma unroll
  for (int k0 = 0; k0 < CN; k0 += 32) {
    v16bf bb = ld16(brow + k0 + boff, brow + k0 + boff + 16);
#pragma unroll
    for (int mi = 0; mi < 4; ++mi) {
      const __bf16* ar = arow + (size_t)(mi * 16) * CN + k0;
      v16bf a = ld16(ar + aoff, ar + 16 + aoff);
      acc[mi] = WMMA_BF16(a, bb, acc[mi]);
    }
  }
  __bf16* orow = OutT + ((size_t)b * SN + (s0 + l15)) * M + m0 + base;
#pragma unroll
  for (int mi = 0; mi < 4; ++mi) {
    v8bf o;
#pragma unroll
    for (int i = 0; i < 8; ++i) o[i] = (__bf16)acc[mi][i];
    *(v8bf*)(orow + mi * 16) = o;
  }
}

// ---------------------------------------------------------------------------
// 4) 3x3 conv (192->192, SAME) as 9 shifted-tap GEMM accumulations.
//    One wave = 64x16 output tile: 9 taps x 6 K-chunks x 4 M-tiles = 216 WMMA.
// ---------------------------------------------------------------------------
__global__ void k_conv3x3(const __bf16* __restrict__ w3t, const __bf16* __restrict__ InT,
                          __bf16* __restrict__ OutT) {
  const int wid  = blockIdx.x * (blockDim.x >> 5) + (threadIdx.x >> 5);
  const int lane = threadIdx.x & 31;
  const int Mblocks = CN / 64, Ntiles = SN / 16;
  if (wid >= BN * Mblocks * Ntiles) return;
  const int b  = wid / (Mblocks * Ntiles);
  const int r  = wid % (Mblocks * Ntiles);
  const int mb = r / Ntiles, nt = r % Ntiles;
  const int m0 = mb * 64, s0 = nt * 16;
  const int l15  = lane & 15;
  const int hi   = lane >> 4;
  const int aoff = hi * 8, boff = hi * 8, base = hi * 8;

  const int s = s0 + l15;
  const int y = s >> 7;          // uniform across the tile
  const int x = s & 127;

  v8f acc[4];
#pragma unroll
  for (int mi = 0; mi < 4; ++mi) acc[mi] = zero8();

#pragma unroll
  for (int t = 0; t < 9; ++t) {
    const int dy = t / 3 - 1, dx = t % 3 - 1;
    const bool valid = ((unsigned)(y + dy) < (unsigned)HN) && ((unsigned)(x + dx) < (unsigned)WN);
    const int sp = valid ? (s + dy * WN + dx) : s;
    const __bf16* arow = w3t + ((size_t)(t * CN + m0 + l15)) * CN;
    const __bf16* brow = InT + ((size_t)b * SN + sp) * CN;
#pragma unroll
    for (int k0 = 0; k0 < CN; k0 += 32) {
      v16bf bb = ld16(brow + k0 + boff, brow + k0 + boff + 16);
      if (!valid) bb = zero16();
#pragma unroll
      for (int mi = 0; mi < 4; ++mi) {
        const __bf16* ar = arow + (size_t)(mi * 16) * CN + k0;
        v16bf a = ld16(ar + aoff, ar + 16 + aoff);
        acc[mi] = WMMA_BF16(a, bb, acc[mi]);
      }
    }
  }
  __bf16* orow = OutT + ((size_t)b * SN + (s0 + l15)) * CN + m0 + base;
#pragma unroll
  for (int mi = 0; mi < 4; ++mi) {
    v8bf o;
#pragma unroll
    for (int i = 0; i < 8; ++i) o[i] = (__bf16)acc[mi][i];
    *(v8bf*)(orow + mi * 16) = o;
  }
}

// ---------------------------------------------------------------------------
// 5) depthwise 3x3 over qkv1T[b][s][c3]; routes q,k -> natural [b][c][s], v -> vT[b][s][c]
// ---------------------------------------------------------------------------
__global__ void k_dwconv(const __bf16* __restrict__ qkv1T, const float* __restrict__ wdw,
                         __bf16* __restrict__ qb, __bf16* __restrict__ kb,
                         __bf16* __restrict__ vT) {
  const size_t tid = (size_t)blockIdx.x * blockDim.x + threadIdx.x;
  if (tid >= (size_t)BN * SN * C3N) return;
  const int c3 = (int)(tid % C3N);
  const size_t t2 = tid / C3N;
  const int s = (int)(t2 % SN);
  const int b = (int)(t2 / SN);
  const int x = s & 127, y = s >> 7;
  float acc = 0.0f;
#pragma unroll
  for (int ky = 0; ky < 3; ++ky)
#pragma unroll
    for (int kx = 0; kx < 3; ++kx) {
      const int yy = y + ky - 1, xx = x + kx - 1;
      if ((unsigned)yy < (unsigned)HN && (unsigned)xx < (unsigned)WN) {
        const int sp = yy * WN + xx;
        acc += (float)qkv1T[((size_t)b * SN + sp) * C3N + c3] * wdw[c3 * 9 + ky * 3 + kx];
      }
    }
  if (c3 < CN)            qb[((size_t)b * CN + c3) * SN + s]        = (__bf16)acc;
  else if (c3 < 2 * CN)   kb[((size_t)b * CN + (c3 - CN)) * SN + s] = (__bf16)acc;
  else                    vT[((size_t)b * SN + s) * CN + (c3 - 2 * CN)] = (__bf16)acc;
}

// ---------------------------------------------------------------------------
// 6) per-(b,c) sum of squares over S (natural: stride 1; transposed: stride CN)
// ---------------------------------------------------------------------------
__global__ void k_sumsq(const __bf16* __restrict__ src, float* __restrict__ out, int natural) {
  __shared__ float sm[256];
  const int row = blockIdx.x;                // b*CN + c
  const int b = row / CN, c = row % CN;
  const size_t basei = natural ? ((size_t)row * SN) : ((size_t)b * SN * CN + c);
  const size_t stride = natural ? 1 : CN;
  float sum = 0.0f;
  for (int i = threadIdx.x; i < SN; i += 256) {
    float v = (float)src[basei + (size_t)i * stride];
    sum += v * v;
  }
  sm[threadIdx.x] = sum; __syncthreads();
  for (int o = 128; o > 0; o >>= 1) {
    if ((int)threadIdx.x < o) sm[threadIdx.x] += sm[threadIdx.x + o];
    __syncthreads();
  }
  if (threadIdx.x == 0) out[row] = sm[0];
}

// ---------------------------------------------------------------------------
// 7) p = q/||q|| + cn_f/||cn_f||   (bf16, natural layout) + sum(p^2)
// ---------------------------------------------------------------------------
__global__ void k_ppass(const __bf16* __restrict__ qb, const __bf16* __restrict__ cnfT,
                        const float* __restrict__ ssq, const float* __restrict__ sscn,
                        __bf16* __restrict__ pb, float* __restrict__ ssp) {
  __shared__ float sm[256];
  const int row = blockIdx.x;                // b*CN + c
  const int b = row / CN, c = row % CN;
  const float inq  = 1.0f / fmaxf(sqrtf(ssq[row]),  EPSN);
  const float incn = 1.0f / fmaxf(sqrtf(sscn[row]), EPSN);
  float sum = 0.0f;
  for (int s = threadIdx.x; s < SN; s += 256) {
    float pv = (float)qb[(size_t)row * SN + s] * inq +
               (float)cnfT[((size_t)b * SN + s) * CN + c] * incn;
    pb[(size_t)row * SN + s] = (__bf16)pv;
    sum += pv * pv;
  }
  sm[threadIdx.x] = sum; __syncthreads();
  for (int o = 128; o > 0; o >>= 1) {
    if ((int)threadIdx.x < o) sm[threadIdx.x] += sm[threadIdx.x + o];
    __syncthreads();
  }
  if (threadIdx.x == 0) ssp[row] = sm[0];
}

// ---------------------------------------------------------------------------
// 8) attention logits, 8-way K-split:
//    araw[ks][b,h,c,d] = sum_{s in chunk ks} p[c,s]*k[d,s]
// ---------------------------------------------------------------------------
__global__ void k_attn_gemm(const __bf16* __restrict__ pb, const __bf16* __restrict__ kb,
                            float* __restrict__ araw) {
  const int wid  = blockIdx.x * (blockDim.x >> 5) + (threadIdx.x >> 5);
  const int lane = threadIdx.x & 31;
  if (wid >= BN * NHD * 9 * KSPL) return;
  const int b = wid / (NHD * 9 * KSPL);
  int r = wid % (NHD * 9 * KSPL);
  const int h  = r / (9 * KSPL); r %= 9 * KSPL;
  const int tt = r / KSPL;
  const int ks = r % KSPL;
  const int ti = tt / 3, tj = tt % 3;
  const int l15  = lane & 15;
  const int hi   = lane >> 4;
  const int aoff = hi * 8, boff = hi * 8, base = hi * 8;

  const int crow = h * CHN + ti * 16 + l15;
  const int dcol = h * CHN + tj * 16 + l15;
  const __bf16* arow = pb + ((size_t)(b * CN + crow)) * SN;
  const __bf16* brow = kb + ((size_t)(b * CN + dcol)) * SN;

  const int kbeg = ks * (SN / KSPL);
  const int kend = kbeg + (SN / KSPL);

  v8f acc = zero8();
  for (int k0 = kbeg; k0 < kend; k0 += 32) {
    v16bf a  = ld16(arow + k0 + aoff, arow + k0 + 16 + aoff);
    v16bf bb = ld16(brow + k0 + boff, brow + k0 + boff + 16);
    acc = WMMA_BF16(a, bb, acc);
  }
  const size_t rows = (size_t)BN * NHD * CHN;
#pragma unroll
  for (int i = 0; i < 8; ++i)
    araw[((size_t)ks * rows + (b * NHD + h) * CHN + ti * 16 + base + i) * CHN +
         tj * 16 + l15] = acc[i];
}

// ---------------------------------------------------------------------------
// 9) reduce K-split partials, scale by temperature/(||p|| ||k||), softmax over d
// ---------------------------------------------------------------------------
__global__ void k_softmax(const float* __restrict__ araw, const float* __restrict__ ssp,
                          const float* __restrict__ ssk, const float* __restrict__ temp,
                          __bf16* __restrict__ attn_bf) {
  const int idx = blockIdx.x * blockDim.x + threadIdx.x;
  if (idx >= BN * NHD * CHN) return;
  const int c = idx % CHN;
  const int h = (idx / CHN) % NHD;
  const int b = idx / (CHN * NHD);
  const float t  = temp[h];
  const float np = fmaxf(sqrtf(ssp[b * CN + h * CHN + c]), EPSN);
  const size_t rows = (size_t)BN * NHD * CHN;
  const size_t row  = (size_t)(b * NHD + h) * CHN + c;
  float lg[CHN];
  float mx = -3.402823466e38f;
#pragma unroll 4
  for (int d = 0; d < CHN; ++d) {
    float v = 0.0f;
#pragma unroll
    for (int ks = 0; ks < KSPL; ++ks) v += araw[((size_t)ks * rows + row) * CHN + d];
    const float nk = fmaxf(sqrtf(ssk[b * CN + h * CHN + d]), EPSN);
    lg[d] = v * t / (np * nk);
    mx = fmaxf(mx, lg[d]);
  }
  float sum = 0.0f;
#pragma unroll 4
  for (int d = 0; d < CHN; ++d) { lg[d] = __expf(lg[d] - mx); sum += lg[d]; }
  const float inv = 1.0f / sum;
  __bf16* orow = attn_bf + row * CHN;
#pragma unroll 4
  for (int d = 0; d < CHN; ++d) orow[d] = (__bf16)(lg[d] * inv);
}

// ---------------------------------------------------------------------------
// 10) outT[s][c] = sum_d vT[s][d] * attn[c][d]   (K = 48 = 32 + half-zero 16)
// ---------------------------------------------------------------------------
__global__ void k_av_gemm(const __bf16* __restrict__ vT, const __bf16* __restrict__ attn_bf,
                          __bf16* __restrict__ outT) {
  const int wid  = blockIdx.x * (blockDim.x >> 5) + (threadIdx.x >> 5);
  const int lane = threadIdx.x & 31;
  const int stiles = SN / 16;
  if (wid >= BN * NHD * 3 * stiles) return;
  const int b = wid / (NHD * 3 * stiles);
  int r = wid % (NHD * 3 * stiles);
  const int h  = r / (3 * stiles); r %= 3 * stiles;
  const int cj = r / stiles;
  const int st = r % stiles;
  const int m0 = st * 16;
  const int l15  = lane & 15;
  const int hi   = lane >> 4;
  const int aoff = hi * 8, base = hi * 8;

  const int srow = m0 + l15;
  const __bf16* va   = vT + ((size_t)b * SN + srow) * CN + h * CHN;
  const int c_local  = cj * 16 + l15;
  const __bf16* brow = attn_bf + ((size_t)((b * NHD + h) * CHN + c_local)) * CHN;

  v8f acc = zero8();
  // chunk 0: d = 0..31 fully valid
  {
    v16bf a  = ld16(va + aoff, va + 16 + aoff);
    v16bf bb = ld16(brow + hi * 8, brow + hi * 8 + 16);
    acc = WMMA_BF16(a, bb, acc);
  }
  // chunk 1: d = 32..47 valid, 48..63 zero-padded
  {
    v16bf a;
    v8bf lo = *(const v8bf*)(va + 32 + aoff);
#pragma unroll
    for (int i = 0; i < 8; ++i) { a[i] = lo[i]; a[i + 8] = (__bf16)0.0f; }
    v8bf blo = *(const v8bf*)(brow + 32 + aoff);
    v16bf bb;
#pragma unroll
    for (int i = 0; i < 8; ++i) { bb[i] = blo[i]; bb[i + 8] = (__bf16)0.0f; }
    acc = WMMA_BF16(a, bb, acc);
  }
  const int cg = h * CHN + cj * 16 + l15;
#pragma unroll
  for (int i = 0; i < 8; ++i)
    outT[((size_t)b * SN + m0 + base + i) * CN + cg] = (__bf16)acc[i];
}

// ---------------------------------------------------------------------------
// 11) proj 1x1 GEMM (4 M-tiles/wave), f32 natural output:
//     out[b][m][s] = sum_c Wp[m][c]*outT[s][c]
// ---------------------------------------------------------------------------
__global__ void k_proj(const __bf16* __restrict__ Wp, const __bf16* __restrict__ outT,
                       float* __restrict__ out) {
  const int wid  = blockIdx.x * (blockDim.x >> 5) + (threadIdx.x >> 5);
  const int lane = threadIdx.x & 31;
  const int Mblocks = CN / 64, Ntiles = SN / 16;
  if (wid >= BN * Mblocks * Ntiles) return;
  const int b  = wid / (Mblocks * Ntiles);
  const int r  = wid % (Mblocks * Ntiles);
  const int mb = r / Ntiles, nt = r % Ntiles;
  const int m0 = mb * 64, s0 = nt * 16;
  const int l15  = lane & 15;
  const int hi   = lane >> 4;
  const int aoff = hi * 8, boff = hi * 8, base = hi * 8;

  const __bf16* arow = Wp + (size_t)(m0 + l15) * CN;
  const __bf16* brow = outT + ((size_t)b * SN + (s0 + l15)) * CN;

  v8f acc[4];
#pragma unroll
  for (int mi = 0; mi < 4; ++mi) acc[mi] = zero8();

#pragma unroll
  for (int k0 = 0; k0 < CN; k0 += 32) {
    v16bf bb = ld16(brow + k0 + boff, brow + k0 + boff + 16);
#pragma unroll
    for (int mi = 0; mi < 4; ++mi) {
      const __bf16* ar = arow + (size_t)(mi * 16) * CN + k0;
      v16bf a = ld16(ar + aoff, ar + 16 + aoff);
      acc[mi] = WMMA_BF16(a, bb, acc[mi]);
    }
  }
#pragma unroll
  for (int mi = 0; mi < 4; ++mi)
#pragma unroll
    for (int i = 0; i < 8; ++i)
      out[((size_t)b * CN + m0 + mi * 16 + base + i) * SN + (s0 + l15)] = acc[mi][i];
}

// ---------------------------------------------------------------------------
// Host-side orchestration
// ---------------------------------------------------------------------------
extern "C" void kernel_launch(void* const* d_in, const int* in_sizes, int n_in,
                              void* d_out, int out_size, void* d_ws, size_t ws_size,
                              hipStream_t stream) {
  (void)in_sizes; (void)n_in; (void)out_size; (void)ws_size;
  const float* x     = (const float*)d_in[0];
  const float* cn    = (const float*)d_in[1];
  const float* w1f   = (const float*)d_in[2];
  const float* w3f   = (const float*)d_in[3];
  const float* wqkvf = (const float*)d_in[4];
  const float* wdwf  = (const float*)d_in[5];
  const float* wprjf = (const float*)d_in[6];
  const float* temp  = (const float*)d_in[7];
  float* out = (float*)d_out;

  const size_t E = (size_t)BN * SN * CN;   // 25,165,824 bf16 elems
  __bf16* wsb = (__bf16*)d_ws;
  // arena with lifetime reuse (peak ~356 MB):
  __bf16* cnT   = wsb + 0;       __bf16* qb   = cnT;    // q overwrites cnT
  __bf16* xT    = wsb + E;       __bf16* kb   = xT;     // k overwrites xT
  __bf16* cn1T  = wsb + 2 * E;   __bf16* vT   = cn1T;   // vT overwrites cn1T
  __bf16* cnfT  = wsb + 3 * E;
  __bf16* qkv1T = wsb + 4 * E;   // 3E, dead after dwconv
  __bf16* pb    = wsb + 4 * E;   // reuse
  __bf16* outT  = wsb + 5 * E;   // reuse
  __bf16* w1b   = wsb + 7 * E;
  __bf16* wqkvb = w1b + CN * CN;
  __bf16* wprjb = wqkvb + C3N * CN;
  __bf16* w3tb  = wprjb + CN * CN;
  __bf16* attnb = w3tb + 9 * CN * CN;
  float* fbase  = (float*)(attnb + BN * NHD * CHN * CHN);
  float* ssq    = fbase;
  float* ssk    = fbase + BN * CN;
  float* sscn   = fbase + 2 * BN * CN;
  float* ssp    = fbase + 3 * BN * CN;
  float* araw   = fbase + 4 * BN * CN;   // KSPL * BN*NHD*CHN*CHN floats (~2.4 MB)

  const dim3 tb(32, 8), tg(SN / 32, CN / 32, BN);
  hipLaunchKernelGGL(k_transpose_bf16, tg, tb, 0, stream, cn, cnT);
  hipLaunchKernelGGL(k_transpose_bf16, tg, tb, 0, stream, x, xT);

  hipLaunchKernelGGL(k_cvt, dim3((CN * CN + 255) / 256), dim3(256), 0, stream, w1f, w1b, CN * CN);
  hipLaunchKernelGGL(k_cvt, dim3((C3N * CN + 255) / 256), dim3(256), 0, stream, wqkvf, wqkvb, C3N * CN);
  hipLaunchKernelGGL(k_cvt, dim3((CN * CN + 255) / 256), dim3(256), 0, stream, wprjf, wprjb, CN * CN);
  hipLaunchKernelGGL(k_cvt_w3t, dim3((9 * CN * CN + 255) / 256), dim3(256), 0, stream, w3f, w3tb);

  // cn branch: 1x1 then 3x3 (all WMMA, 64-row M blocks)
  hipLaunchKernelGGL(k_gemm_nt, dim3(BN * (CN / 64) * (SN / 16) / 8), dim3(256), 0, stream,
                     w1b, cnT, cn1T, CN, CN / 64);
  hipLaunchKernelGGL(k_conv3x3, dim3(BN * (CN / 64) * (SN / 16) / 8), dim3(256), 0, stream,
                     w3tb, cn1T, cnfT);

  // qkv: 1x1 (WMMA) then depthwise 3x3
  hipLaunchKernelGGL(k_gemm_nt, dim3(BN * (C3N / 64) * (SN / 16) / 8), dim3(256), 0, stream,
                     wqkvb, xT, qkv1T, C3N, C3N / 64);
  {
    size_t tot = (size_t)BN * SN * C3N;
    hipLaunchKernelGGL(k_dwconv, dim3((unsigned)((tot + 255) / 256)), dim3(256), 0, stream,
                       qkv1T, wdwf, qb, kb, vT);
  }

  // L2 norms (sum of squares) for q, k, cn_f
  hipLaunchKernelGGL(k_sumsq, dim3(BN * CN), dim3(256), 0, stream, qb, ssq, 1);
  hipLaunchKernelGGL(k_sumsq, dim3(BN * CN), dim3(256), 0, stream, kb, ssk, 1);
  hipLaunchKernelGGL(k_sumsq, dim3(BN * CN), dim3(256), 0, stream, cnfT, sscn, 0);

  // p = l2n(q) + l2n(cn_f), plus ||p||^2
  hipLaunchKernelGGL(k_ppass, dim3(BN * CN), dim3(256), 0, stream, qb, cnfT, ssq, sscn, pb, ssp);

  // channel attention (K-split logits GEMM, reduction folded into softmax)
  hipLaunchKernelGGL(k_attn_gemm, dim3(BN * NHD * 9 * KSPL / 8), dim3(256), 0, stream,
                     pb, kb, araw);
  hipLaunchKernelGGL(k_softmax, dim3((BN * NHD * CHN + 255) / 256), dim3(256), 0, stream,
                     araw, ssp, ssk, temp, attnb);
  hipLaunchKernelGGL(k_av_gemm, dim3(BN * NHD * 3 * (SN / 16) / 8), dim3(256), 0, stream,
                     vT, attnb, outT);

  // final 1x1 projection -> f32 output
  hipLaunchKernelGGL(k_proj, dim3(BN * (CN / 64) * (SN / 16) / 8), dim3(256), 0, stream,
                     wprjb, outT, out);
}
